// Bitcodes_33311766347986
// MI455X (gfx1250) — matrix operand order repaired and checked by reading, not compiled, not run on hardware
//
#include <hip/hip_runtime.h>

// Problem constants (from reference)
#define RROWS 32768   // B*M
#define DDIM  1024    // D (= K of GEMM1, N of GEMM2)
#define NCB   1024    // codebook rows (= N of GEMM1)
#define PPAIR 512     // NUM_BITS (= K of GEMM2)

typedef __attribute__((ext_vector_type(16))) __bf16 vbf16;
typedef __attribute__((ext_vector_type(8)))  __bf16 vbf8;
typedef __attribute__((ext_vector_type(8)))  float  v8f;

// ---------------------------------------------------------- CDNA5 async path

// LDS byte offset of a generic pointer to __shared__ memory (addrspace(3)).
__device__ __forceinline__ unsigned lds_off(const void* p) {
  return (unsigned)(unsigned long long)(const __attribute__((address_space(3))) char*)p;
}

// GLOBAL_LOAD_ASYNC_TO_LDS_B128: 16 bytes per lane, global -> LDS, ASYNCcnt.
__device__ __forceinline__ void async_b128(const void* gsrc, const void* ldsdst) {
  asm volatile("global_load_async_to_lds_b128 %0, %1, off"
               :: "v"(lds_off(ldsdst)), "v"(gsrc)
               : "memory");
}

__device__ __forceinline__ void wait_async(int n) {
  asm volatile("s_wait_asynccnt %0" :: "n"(n) : "memory");
}

// ---------------------------------------------------------------- prep kernels

__global__ __launch_bounds__(256) void cvt_f32_bf16_kernel(
    const float* __restrict__ src, __bf16* __restrict__ dst, int n) {
  int i = (blockIdx.x * 256 + threadIdx.x) * 8;
  if (i + 8 <= n) {
    float4 a = *(const float4*)(src + i);
    float4 b = *(const float4*)(src + i + 4);
    vbf8 v;
    v[0] = (__bf16)a.x; v[1] = (__bf16)a.y; v[2] = (__bf16)a.z; v[3] = (__bf16)a.w;
    v[4] = (__bf16)b.x; v[5] = (__bf16)b.y; v[6] = (__bf16)b.z; v[7] = (__bf16)b.w;
    *(vbf8*)(dst + i) = v;
  }
}

// base[d] = sum_p codebook[2p, d]
__global__ __launch_bounds__(256) void base_kernel(
    const float* __restrict__ cb, float* __restrict__ base) {
  int d = blockIdx.x * 256 + threadIdx.x;
  if (d < DDIM) {
    float s = 0.f;
    for (int p = 0; p < PPAIR; ++p) s += cb[(size_t)(2 * p) * DDIM + d];
    base[d] = s;
  }
}

// deltaT[d, p] = bf16(codebook[2p+1,d] - codebook[2p,d])  (transposed: [DDIM][PPAIR])
__global__ __launch_bounds__(256) void deltaT_kernel(
    const float* __restrict__ cb, __bf16* __restrict__ deltaT) {
  int i = blockIdx.x * 256 + threadIdx.x;   // d fast for coalesced reads
  if (i < PPAIR * DDIM) {
    int p = i >> 10, d = i & (DDIM - 1);
    float e = cb[(size_t)(2 * p) * DDIM + d];
    float o = cb[(size_t)(2 * p + 1) * DDIM + d];
    deltaT[(size_t)d * PPAIR + p] = (__bf16)(o - e);
  }
}

// ----------------------------------------------------- shared GEMM machinery
// Both GEMMs: C[128x128 tile] = A[rows x K] * B[cols x K]^T (K contiguous in
// both operands). 256 threads = 8 waves; wave computes 32x64 via 2x4 WMMA
// tiles. LDS double-buffered, filled with GLOBAL_LOAD_ASYNC_TO_LDS_B128.

struct Frags { vbf16 a[2]; vbf16 b[4]; };

__device__ __forceinline__ void load_frags(
    const __bf16* As, const __bf16* Bs, int wm0, int wn0, int g, int ml,
    Frags& f) {
#pragma unroll
  for (int t = 0; t < 2; ++t) {     // 16-bit A 16x32: 16 contig K (matches B)
    union { vbf16 v; vbf8 h[2]; } u;
    const int r = (wm0 + t * 16 + ml) * 32;
    u.h[0] = *(const vbf8*)(As + r + g * 16);
    u.h[1] = *(const vbf8*)(As + r + g * 16 + 8);
    f.a[t] = u.v;
  }
#pragma unroll
  for (int u = 0; u < 4; ++u)       // 16-bit B 32x16: 16 contig K (matches A)
    f.b[u] = *(const vbf16*)(Bs + (wn0 + u * 16 + ml) * 32 + g * 16);
}

// ------------------------------------------------- GEMM1: sim + pair-argmax

__global__ __launch_bounds__(256) void sim_bits_kernel(
    const __bf16* __restrict__ xb, const __bf16* __restrict__ cb16,
    int* __restrict__ bits, __bf16* __restrict__ bitsf) {
  __shared__ alignas(32) __bf16 As[2][128 * 32];
  __shared__ alignas(32) __bf16 Bs[2][128 * 32];

  const int tid  = threadIdx.x;
  const int lane = tid & 31;
  const int wave = tid >> 5;
  const int rowBase = blockIdx.x * 128;
  const int nBase   = blockIdx.y * 128;
  const int wm0 = (wave >> 1) * 32;
  const int wn0 = (wave & 1) * 64;
  const int g  = lane >> 4;
  const int ml = lane & 15;

  const int ldRow = tid >> 1;         // 0..127
  const int ldK   = (tid & 1) * 16;   // 0 or 16

  const __bf16* gA = xb   + (size_t)(rowBase + ldRow) * DDIM + ldK;
  const __bf16* gB = cb16 + (size_t)(nBase   + ldRow) * DDIM + ldK;
  const int ldsIdx = ldRow * 32 + ldK;

  v8f acc[2][4] = {};

  // prologue: fill buffer 0 (4 async B128s per thread: 2 for A, 2 for B)
  async_b128(gA,     &As[0][ldsIdx]);
  async_b128(gA + 8, &As[0][ldsIdx + 8]);
  async_b128(gB,     &Bs[0][ldsIdx]);
  async_b128(gB + 8, &Bs[0][ldsIdx + 8]);

  int cur = 0;
  for (int k0 = 0; k0 < DDIM; k0 += 32) {
    const int nxt = cur ^ 1;
    const bool more = (k0 + 32) < DDIM;
    if (more) {                        // DMA next slab while computing this one
      async_b128(gA + k0 + 32,     &As[nxt][ldsIdx]);
      async_b128(gA + k0 + 32 + 8, &As[nxt][ldsIdx + 8]);
      async_b128(gB + k0 + 32,     &Bs[nxt][ldsIdx]);
      async_b128(gB + k0 + 32 + 8, &Bs[nxt][ldsIdx + 8]);
      wait_async(4);                   // in-order: <=4 outstanding => cur done
    } else {
      wait_async(0);
    }
    __syncthreads();

    Frags f;
    load_frags(As[cur], Bs[cur], wm0, wn0, g, ml, f);
#pragma unroll
    for (int t = 0; t < 2; ++t)
#pragma unroll
      for (int u = 0; u < 4; ++u)
        acc[t][u] = __builtin_amdgcn_wmma_f32_16x16x32_bf16(
            false, f.a[t], false, f.b[u], (short)0, acc[t][u], false, false);
    __syncthreads();                   // all waves done before cur is refilled
    cur = nxt;
  }

  // Epilogue: adjacent N columns sit in adjacent lanes of the C layout.
#pragma unroll
  for (int t = 0; t < 2; ++t) {
#pragma unroll
    for (int u = 0; u < 4; ++u) {
      const int nglob = nBase + wn0 + u * 16 + ml;
#pragma unroll
      for (int v = 0; v < 8; ++v) {
        float s = acc[t][u][v];
        float o = __shfl_xor(s, 1, 32);
        if ((ml & 1) == 0) {           // even lane holds sim[2p]
          int row = rowBase + wm0 + t * 16 + g * 8 + v;
          int bit = (o > s) ? 1 : 0;   // argmax: 1 iff sim[2p+1] > sim[2p]
          size_t bi = (size_t)row * PPAIR + (nglob >> 1);
          bits[bi]  = bit;
          bitsf[bi] = (__bf16)(bit ? 1.0f : 0.0f);
        }
      }
    }
  }
}

// --------------------------------------- GEMM2: out = bitsf @ DeltaT^T + base

__global__ __launch_bounds__(256) void recon_kernel(
    const __bf16* __restrict__ bitsf, const __bf16* __restrict__ deltaT,
    const float* __restrict__ base, float* __restrict__ out) {
  __shared__ alignas(32) __bf16 As[2][128 * 32];
  __shared__ alignas(32) __bf16 Bs[2][128 * 32];

  const int tid  = threadIdx.x;
  const int lane = tid & 31;
  const int wave = tid >> 5;
  const int rowBase = blockIdx.x * 128;
  const int nBase   = blockIdx.y * 128;
  const int wm0 = (wave >> 1) * 32;
  const int wn0 = (wave & 1) * 64;
  const int g  = lane >> 4;
  const int ml = lane & 15;

  const int ldRow = tid >> 1;
  const int ldK   = (tid & 1) * 16;

  const __bf16* gA = bitsf  + (size_t)(rowBase + ldRow) * PPAIR + ldK;
  const __bf16* gB = deltaT + (size_t)(nBase   + ldRow) * PPAIR + ldK;
  const int ldsIdx = ldRow * 32 + ldK;

  v8f acc[2][4] = {};

  async_b128(gA,     &As[0][ldsIdx]);
  async_b128(gA + 8, &As[0][ldsIdx + 8]);
  async_b128(gB,     &Bs[0][ldsIdx]);
  async_b128(gB + 8, &Bs[0][ldsIdx + 8]);

  int cur = 0;
  for (int k0 = 0; k0 < PPAIR; k0 += 32) {
    const int nxt = cur ^ 1;
    const bool more = (k0 + 32) < PPAIR;
    if (more) {
      async_b128(gA + k0 + 32,     &As[nxt][ldsIdx]);
      async_b128(gA + k0 + 32 + 8, &As[nxt][ldsIdx + 8]);
      async_b128(gB + k0 + 32,     &Bs[nxt][ldsIdx]);
      async_b128(gB + k0 + 32 + 8, &Bs[nxt][ldsIdx + 8]);
      wait_async(4);
    } else {
      wait_async(0);
    }
    __syncthreads();

    Frags f;
    load_frags(As[cur], Bs[cur], wm0, wn0, g, ml, f);
#pragma unroll
    for (int t = 0; t < 2; ++t)
#pragma unroll
      for (int u = 0; u < 4; ++u)
        acc[t][u] = __builtin_amdgcn_wmma_f32_16x16x32_bf16(
            false, f.a[t], false, f.b[u], (short)0, acc[t][u], false, false);
    __syncthreads();
    cur = nxt;
  }

#pragma unroll
  for (int t = 0; t < 2; ++t) {
#pragma unroll
    for (int u = 0; u < 4; ++u) {
      const int nglob = nBase + wn0 + u * 16 + ml;
      const float bz = base[nglob];
#pragma unroll
      for (int v = 0; v < 8; ++v) {
        int row = rowBase + wm0 + t * 16 + g * 8 + v;
        out[(size_t)row * DDIM + nglob] = acc[t][u][v] + bz;
      }
    }
  }
}

// ---------------------------------------------------------------- launcher

extern "C" void kernel_launch(void* const* d_in, const int* in_sizes, int n_in,
                              void* d_out, int out_size, void* d_ws, size_t ws_size,
                              hipStream_t stream) {
  const float* x  = (const float*)d_in[0];   // [8,4096,1024] f32
  const float* cb = (const float*)d_in[1];   // [1024,1024]  f32

  float* out  = (float*)d_out;                       // [R, D] f32
  int*   bits = (int*)(out + (size_t)RROWS * DDIM);  // [R, P] i32

  char* ws = (char*)d_ws;
  size_t o_xb    = 0;
  size_t o_cb    = o_xb    + (size_t)RROWS * DDIM * 2;
  size_t o_delta = o_cb    + (size_t)NCB   * DDIM * 2;
  size_t o_base  = o_delta + (size_t)PPAIR * DDIM * 2;
  size_t o_bitsf = o_base  + (size_t)DDIM * 4;

  __bf16* xb     = (__bf16*)(ws + o_xb);
  __bf16* cb16   = (__bf16*)(ws + o_cb);
  __bf16* deltaT = (__bf16*)(ws + o_delta);
  float*  basep  = (float*)(ws + o_base);
  __bf16* bitsfp = (__bf16*)(ws + o_bitsf);

  {
    int n = RROWS * DDIM;
    cvt_f32_bf16_kernel<<<(n / 8 + 255) / 256, 256, 0, stream>>>(x, xb, n);
  }
  {
    int n = NCB * DDIM;
    cvt_f32_bf16_kernel<<<(n / 8 + 255) / 256, 256, 0, stream>>>(cb, cb16, n);
  }
  base_kernel<<<(DDIM + 255) / 256, 256, 0, stream>>>(cb, basep);
  deltaT_kernel<<<(PPAIR * DDIM + 255) / 256, 256, 0, stream>>>(cb, deltaT);

  dim3 gemmGrid(RROWS / 128, NCB / 128);  // 256 x 8
  sim_bits_kernel<<<gemmGrid, 256, 0, stream>>>(xb, cb16, bits, bitsfp);

  dim3 reconGrid(RROWS / 128, DDIM / 128);  // 256 x 8
  recon_kernel<<<reconGrid, 256, 0, stream>>>(bitsfp, deltaT, basep, out);
}
